// GNNEncoder_70007966925398
// MI455X (gfx1250) — compile-verified, matrix operand order checked by reading, not run on hardware
//
#include <hip/hip_runtime.h>
#include <hip/hip_bf16.h>
#include <stdint.h>

#define HID  64
#define OUTF 128
#define EPSL 1e-5f

typedef __attribute__((ext_vector_type(16))) _Float16 v16h;
typedef __attribute__((ext_vector_type(8)))  _Float16 v8h;
typedef __attribute__((ext_vector_type(8)))  float    v8f;

// ---------------- degree / normalization ----------------
__global__ void k_deg_init(float* __restrict__ deg, int N) {
  int i = blockIdx.x * blockDim.x + threadIdx.x;
  if (i < N) deg[i] = 1.0f;                       // self-loop contributes 1
}

__global__ void k_deg_accum(const int* __restrict__ dst, float* __restrict__ deg, int E) {
  int e = blockIdx.x * blockDim.x + threadIdx.x;
  if (e < E) atomicAdd(&deg[dst[e]], 1.0f);
}

__global__ void k_deg_to_dinv(float* __restrict__ deg, int N) {
  int i = blockIdx.x * blockDim.x + threadIdx.x;
  if (i < N) deg[i] = rsqrtf(fmaxf(deg[i], 1.0f));
}

__global__ void k_zero(float* __restrict__ p, long long n) {
  long long i = (long long)blockIdx.x * blockDim.x + threadIdx.x;
  if (i < n) p[i] = 0.0f;
}

// ---------------- dense feature transform: out[N x 64] = A[N x K] @ W[K x 64] ----------------
// Block = 128 threads = 4 wave32s; block tile = 64 rows x 64 cols.
// Wave w owns column tile w and iterates 4 row tiles, holding the B fragment in
// registers across them (4x B reuse). A (64 x K) and W^T (64 x K) staged in LDS
// as f16 with padded stride so every fragment piece is one ds_load_b128.
template <int K>
__global__ void __launch_bounds__(128)
k_gemm_wmma(const float* __restrict__ A, const float* __restrict__ W,
            float* __restrict__ out, int N) {
  constexpr int STR = K + 8;                      // pad (multiple of 8): conflict-free, b128-aligned
  __shared__ alignas(16) _Float16 sA[64 * STR];   // A rows
  __shared__ alignas(16) _Float16 sWt[HID * STR]; // W transposed: sWt[col][k]

  const int tid  = threadIdx.x;
  const int row0 = blockIdx.x * 64;

  // stage W (row-major K x 64) transposed as f16
  for (int i = tid; i < K * HID; i += 128) {
    int k = i >> 6, c = i & 63;                   // HID == 64
    sWt[c * STR + k] = (_Float16)W[i];
  }
  // stage the 64 x K A-tile as f16 (clamp rows for the final partial tile)
  for (int i = tid; i < 64 * K; i += 128) {
    int r = i / K, c = i - r * K;
    int rr = row0 + r; if (rr >= N) rr = N - 1;
    sA[r * STR + c] = (_Float16)A[(size_t)rr * K + c];
  }
  __syncthreads();

  const int lane = tid & 31;
  const int wave = tid >> 5;
  const int mrow = lane & 15;
  const int koff = (lane & 16) ? 8 : 0;           // ISA 16-bit A/B fragment layout
  const int col  = wave * 16 + (lane & 15);

  v8f acc0 = {}, acc1 = {}, acc2 = {}, acc3 = {};
#pragma unroll
  for (int k0 = 0; k0 < K; k0 += 32) {
    const int kb = k0 + koff;
    // B fragment: two aligned 16B chunks from transposed W
    v8h blo = *(const v8h*)&sWt[col * STR + kb];
    v8h bhi = *(const v8h*)&sWt[col * STR + kb + 16];
    v16h b = __builtin_shufflevector(blo, bhi, 0, 1, 2, 3, 4, 5, 6, 7,
                                     8, 9, 10, 11, 12, 13, 14, 15);
#pragma unroll
    for (int rt = 0; rt < 4; ++rt) {
      v8h alo = *(const v8h*)&sA[(rt * 16 + mrow) * STR + kb];
      v8h ahi = *(const v8h*)&sA[(rt * 16 + mrow) * STR + kb + 16];
      v16h a = __builtin_shufflevector(alo, ahi, 0, 1, 2, 3, 4, 5, 6, 7,
                                       8, 9, 10, 11, 12, 13, 14, 15);
      v8f c = (rt == 0) ? acc0 : (rt == 1) ? acc1 : (rt == 2) ? acc2 : acc3;
      c = __builtin_amdgcn_wmma_f32_16x16x32_f16(
          /*neg_a=*/false, a, /*neg_b=*/false, b,
          /*c_mod=*/(short)0, c, /*reuse_a=*/false, /*reuse_b=*/false);
      if (rt == 0) acc0 = c; else if (rt == 1) acc1 = c;
      else if (rt == 2) acc2 = c; else acc3 = c;
    }
  }

  // C/D layout: vgpr r -> row r (lanes 0-15) / row r+8 (lanes 16-31), N = lane&15
  const int rhi = (lane & 16) ? 8 : 0;
#pragma unroll
  for (int rt = 0; rt < 4; ++rt) {
    v8f c = (rt == 0) ? acc0 : (rt == 1) ? acc1 : (rt == 2) ? acc2 : acc3;
#pragma unroll
    for (int r = 0; r < 8; ++r) {
      int row = row0 + rt * 16 + rhi + r;
      if (row < N) out[(size_t)row * HID + col] = c[r];
    }
  }
}

// ---------------- message gather + scatter-add ----------------
// one thread per (edge, feature): coalesced 256B reads of hw[src], f32 atomics into agg[dst]
__global__ void k_edge_aggregate(const int* __restrict__ src, const int* __restrict__ dst,
                                 const float* __restrict__ dinv, const float* __restrict__ hw,
                                 float* __restrict__ agg, int E) {
  long long t = (long long)blockIdx.x * blockDim.x + threadIdx.x;
  long long e = t >> 6;
  int f = (int)(t & 63);
  if (e >= E) return;
  int s = src[e], d = dst[e];
  float w = dinv[s] * dinv[d];
  atomicAdd(&agg[(size_t)d * HID + f], hw[(size_t)s * HID + f] * w);
}

// ---------------- self-loop + bias + LayerNorm + ReLU + residual ----------------
// one wave32 per node, 2 features per lane; wave32-wide shfl reductions
__global__ void __launch_bounds__(256)
k_post_ln_relu(const float* __restrict__ agg, const float* __restrict__ hw,
               const float* __restrict__ dinv, const float* __restrict__ bias,
               const float* __restrict__ g, const float* __restrict__ beta,
               float* __restrict__ h, int N, int residual) {
  int wave = threadIdx.x >> 5;
  int lane = threadIdx.x & 31;
  int node = blockIdx.x * 8 + wave;
  if (node >= N) return;
  size_t base = (size_t)node * HID;
  float di = dinv[node];
  float sl = di * di;                             // self-loop norm dinv[i]^2
  float v0 = agg[base + lane]      + hw[base + lane]      * sl + bias[lane];
  float v1 = agg[base + lane + 32] + hw[base + lane + 32] * sl + bias[lane + 32];

  float s = v0 + v1;
#pragma unroll
  for (int m = 16; m >= 1; m >>= 1) s += __shfl_xor(s, m, 32);
  float mean = s * (1.0f / 64.0f);
  float d0 = v0 - mean, d1 = v1 - mean;
  float q = d0 * d0 + d1 * d1;
#pragma unroll
  for (int m = 16; m >= 1; m >>= 1) q += __shfl_xor(q, m, 32);
  float inv = rsqrtf(q * (1.0f / 64.0f) + EPSL);

  float y0 = fmaxf(d0 * inv * g[lane]      + beta[lane],      0.0f);
  float y1 = fmaxf(d1 * inv * g[lane + 32] + beta[lane + 32], 0.0f);
  if (residual) { y0 += h[base + lane]; y1 += h[base + lane + 32]; }
  h[base + lane]      = y0;
  h[base + lane + 32] = y1;
}

// ---------------- mean pool (column sums) ----------------
__global__ void __launch_bounds__(256)
k_pool_sum(const float* __restrict__ h, float* __restrict__ pooled, int N) {
  __shared__ float sacc[HID];
  if (threadIdx.x < HID) sacc[threadIdx.x] = 0.0f;
  __syncthreads();
  int f = threadIdx.x & 63;
  int nloc = threadIdx.x >> 6;                    // 4 nodes per block per pass
  float acc = 0.0f;
  for (long long node = (long long)blockIdx.x * 4 + nloc; node < N;
       node += (long long)gridDim.x * 4)
    acc += h[node * HID + f];
  atomicAdd(&sacc[f], acc);
  __syncthreads();
  if (threadIdx.x < HID) atomicAdd(&pooled[threadIdx.x], sacc[threadIdx.x]);
}

// ---------------- MLP head on the pooled vector (1x64 -> 1x64 -> 1x128) ----------------
__global__ void __launch_bounds__(128)
k_head(const float* __restrict__ pooled, const float* __restrict__ mW0,
       const float* __restrict__ mb0, const float* __restrict__ mW1,
       const float* __restrict__ mb1, float* __restrict__ out, int N) {
  __shared__ float hg[HID];
  __shared__ float hid[HID];
  int t = threadIdx.x;
  if (t < HID) hg[t] = pooled[t] / (float)N;
  __syncthreads();
  if (t < HID) {
    float a = mb0[t];
    for (int k = 0; k < HID; ++k) a += hg[k] * mW0[k * HID + t];
    hid[t] = fmaxf(a, 0.0f);
  }
  __syncthreads();
  float a = mb1[t];
  for (int j = 0; j < HID; ++j) a += hid[j] * mW1[j * OUTF + t];
  out[t] = a;
}

extern "C" void kernel_launch(void* const* d_in, const int* in_sizes, int n_in,
                              void* d_out, int out_size, void* d_ws, size_t ws_size,
                              hipStream_t stream) {
  const float* x   = (const float*)d_in[0];
  const int*   ei  = (const int*)d_in[1];
  const float* W0  = (const float*)d_in[2];  const float* b0  = (const float*)d_in[3];
  const float* g0  = (const float*)d_in[4];  const float* be0 = (const float*)d_in[5];
  const float* W1  = (const float*)d_in[6];  const float* b1  = (const float*)d_in[7];
  const float* g1  = (const float*)d_in[8];  const float* be1 = (const float*)d_in[9];
  const float* W2  = (const float*)d_in[10]; const float* b2  = (const float*)d_in[11];
  const float* g2  = (const float*)d_in[12]; const float* be2 = (const float*)d_in[13];
  const float* mW0 = (const float*)d_in[14]; const float* mb0 = (const float*)d_in[15];
  const float* mW1 = (const float*)d_in[16]; const float* mb1 = (const float*)d_in[17];

  const int N = in_sizes[0] / 128;   // 50000
  const int E = in_sizes[1] / 2;     // 800000
  const int* src = ei;
  const int* dst = ei + E;

  // workspace layout (floats), 256B-aligned sections
  float* ws     = (float*)d_ws;
  size_t o      = 0;
  float* dinv   = ws + o; o += ((size_t)N + 63) & ~(size_t)63;
  float* h      = ws + o; o += (size_t)N * HID;
  float* hw     = ws + o; o += (size_t)N * HID;
  float* agg    = ws + o; o += (size_t)N * HID;
  float* pooled = ws + o;

  // --- degree / symmetric normalization ---
  k_deg_init   <<<(N + 255) / 256, 256, 0, stream>>>(dinv, N);
  k_deg_accum  <<<(E + 255) / 256, 256, 0, stream>>>(dst, dinv, E);
  k_deg_to_dinv<<<(N + 255) / 256, 256, 0, stream>>>(dinv, N);

  const int mtiles = (N + 63) / 64;
  const long long nh = (long long)N * HID;
  const long long ef = (long long)E * HID;

  const float* Wl[3]  = {W0, W1, W2};
  const float* bl[3]  = {b0, b1, b2};
  const float* gl[3]  = {g0, g1, g2};
  const float* bel[3] = {be0, be1, be2};

  for (int L = 0; L < 3; ++L) {
    // hw = h @ W  (WMMA)
    if (L == 0)
      k_gemm_wmma<128><<<mtiles, 128, 0, stream>>>(x, Wl[L], hw, N);
    else
      k_gemm_wmma<64> <<<mtiles, 128, 0, stream>>>(h, Wl[L], hw, N);
    // agg = scatter_add(hw[src] * norm)
    k_zero<<<(int)((nh + 255) / 256), 256, 0, stream>>>(agg, nh);
    k_edge_aggregate<<<(int)((ef + 255) / 256), 256, 0, stream>>>(src, dst, dinv, hw, agg, E);
    // h = [h +] relu(LN(agg + self_loop + b))
    k_post_ln_relu<<<(N + 7) / 8, 256, 0, stream>>>(agg, hw, dinv, bl[L], gl[L], bel[L],
                                                    h, N, L > 0 ? 1 : 0);
  }

  // --- mean pool + MLP head ---
  k_zero<<<1, 64, 0, stream>>>(pooled, HID);
  k_pool_sum<<<256, 256, 0, stream>>>(h, pooled, N);
  k_head<<<1, 128, 0, stream>>>(pooled, mW0, mb0, mW1, mb1, (float*)d_out, N);
}
